// AE_30142080483951
// MI455X (gfx1250) — matrix operand order repaired and checked by reading, not compiled, not run on hardware
//
#include <hip/hip_runtime.h>

typedef __attribute__((ext_vector_type(2)))  float  v2f;
typedef __attribute__((ext_vector_type(8)))  float  v8f;
typedef __attribute__((ext_vector_type(8)))  __bf16 v8bf;
typedef __attribute__((ext_vector_type(16))) __bf16 v16bf;

static constexpr int Bb  = 8;
static constexpr int Nn  = 48;
static constexpr int Mm  = Nn * Nn;   // 2304 tokens
static constexpr int Cs  = 64;
static constexpr int Cin = 256;
static constexpr int Ci  = 128;
static constexpr int Co  = 128;
static constexpr int Hsp = 96;        // SP spatial (2N)
#define EPSV 1e-5f

// ---------------------------------------------------------------------------
// Generic strided batched GEMM on V_WMMA_F32_16X16X4_F32 (wave32).
// C[b](i,j) = epilogue( sum_k A[b](i,k) * B[b](k,j) )
//   epilogue: v = v*scale_m[i] + bias_m[i] + bias_n[j]; relu; v += res[b](i,j)
// One wave computes one 16x16 tile; 4 waves per block cover 64 columns.
// Template AC/BC: K-contiguous operand -> single 8-byte fragment load.
// Software-pipelined: next fragments are in flight during the current wmma.
//
// f32 WMMA fragment layouts (CDNA5 ISA 7.12.2):
//   A 16x4 : lanes 0-15 -> row=lane,  VGPR0/1 = K0,K1 ; lanes16-31 -> K2,K3
//   B 4x16 : lanes 0-15 -> col=lane,  VGPR0/1 = K0,K1 ; lanes16-31 -> K2,K3
//   C 16x16: VGPR p: lanes0-15 -> (M=p, N=lane); lanes16-31 -> (M=8+p, N=lane-16)
// ---------------------------------------------------------------------------
template <bool AC, bool BC>
__global__ void wmma_gemm_f32_kern(
    const float* __restrict__ A, long long As0, long long As1, long long Abat,
    const float* __restrict__ Bm, long long Bs0, long long Bs1, long long Bbat,
    float* __restrict__ C, long long Cs0, long long Cs1, long long Cbat,
    const float* __restrict__ scale_m, const float* __restrict__ bias_m,
    const float* __restrict__ bias_n,
    const float* __restrict__ res, long long Rs0, long long Rs1, long long Rbat,
    int M, int N, int K, int relu)
{
    const int wave = threadIdx.x >> 5;
    const int lane = threadIdx.x & 31;
    const int i0 = blockIdx.y << 4;
    const int j0 = (blockIdx.x * 4 + wave) << 4;
    const int b  = blockIdx.z;
    if (j0 >= N) return;                       // wave-uniform: EXEC stays all-1s

    A  += (long long)b * Abat;
    Bm += (long long)b * Bbat;
    C  += (long long)b * Cbat;

    const int r  = lane & 15;                  // A row / B col / C col
    const int hf = lane >> 4;                  // lane half
    const int kh = hf * 2;                     // K sub-offset for this half

    const float* Abase = A  + (long long)(i0 + r) * As0;
    const float* Bbase = Bm + (long long)(j0 + r) * Bs1;

    auto lda = [&](int k) -> v2f {
        if constexpr (AC) {
            return *(const v2f*)(Abase + k + kh);        // As1 == 1, 8B aligned
        } else {
            const float* Ap = Abase + (long long)(k + kh) * As1;
            v2f a; a.x = Ap[0]; a.y = Ap[As1]; return a;
        }
    };
    auto ldb = [&](int k) -> v2f {
        if constexpr (BC) {
            return *(const v2f*)(Bbase + k + kh);        // Bs0 == 1, 8B aligned
        } else {
            const float* Bp = Bbase + (long long)(k + kh) * Bs0;
            v2f v; v.x = Bp[0]; v.y = Bp[Bs0]; return v;
        }
    };

    v8f acc = {};
    v2f af = lda(0), bf = ldb(0);
    for (int k = 4; k < K; k += 4) {
        v2f af2 = lda(k), bf2 = ldb(k);        // prefetch next fragments
        acc = __builtin_amdgcn_wmma_f32_16x16x4_f32(false, af, false, bf,
                                                    (short)0, acc, false, false);
        af = af2; bf = bf2;
    }
    acc = __builtin_amdgcn_wmma_f32_16x16x4_f32(false, af, false, bf,
                                                (short)0, acc, false, false);

#pragma unroll
    for (int p = 0; p < 8; ++p) {
        const int i = i0 + hf * 8 + p;
        const int j = j0 + r;
        float v = acc[p];
        if (scale_m) v *= scale_m[i];
        if (bias_m)  v += bias_m[i];
        if (bias_n)  v += bias_n[j];
        if (relu)    v  = fmaxf(v, 0.f);
        if (res)     v += res[(long long)b * Rbat + (long long)i * Rs0
                             + (long long)j * Rs1];
        C[(long long)i * Cs0 + (long long)j * Cs1] = v;
    }
}

// ---------------------------------------------------------------------------
// Dominant GEMM: C[b] (M x N, f32) = Asp (M x K bf16, row-major)
//                                    x Y   (B x N x K bf16, channel-major)
// on V_WMMA_F32_16X16X32_BF16.  B(k,j) = Y[b, j, k], so both the 16x32 A
// fragment and the 32x16 B fragments are contiguous 16B loads per lane:
//   A frag (lane half hf, row r): elems 0-7  = A[r][k + 8*hf + 0..7]
//                                 elems 8-15 = A[r][k + 16 + 8*hf + 0..7]
//   B frag (col r):               elems 0-7 = B[k + 8*hf + 0..7][col], 8-15 = B[k + 16 + 8*hf + 0..7][col]
// Register blocking: each wave computes a 16x32 C tile (2 accumulators share
// one A fragment -> halves Asp traffic, doubles wmma per address setup).
// ---------------------------------------------------------------------------
__global__ void wmma_gemm_bf16_kern(
    const __bf16* __restrict__ Asp,   // (M x K) row-major
    const __bf16* __restrict__ Ybf,   // (batch x N x K)
    float* __restrict__ C,            // (batch x M x N)
    int M, int N, int K)
{
    const int wave = threadIdx.x >> 5;
    const int lane = threadIdx.x & 31;
    const int i0 = blockIdx.y << 4;
    const int j0 = (blockIdx.x * 4 + wave) << 5;   // 32 columns per wave
    const int b  = blockIdx.z;
    if (j0 >= N) return;

    const int r  = lane & 15;
    const int hf = lane >> 4;

    const __bf16* Arow = Asp + (long long)(i0 + r) * K;
    const __bf16* B0   = Ybf + ((long long)b * N + (j0 + r)) * K;
    const __bf16* B1   = B0 + (long long)16 * K;

    auto ldA = [&](int k) -> v16bf {
        v8bf lo = *(const v8bf*)(Arow + k + 8 * hf);
        v8bf hi = *(const v8bf*)(Arow + k + 16 + 8 * hf);
        return __builtin_shufflevector(lo, hi, 0,1,2,3,4,5,6,7,
                                               8,9,10,11,12,13,14,15);
    };
    auto ldB = [&](const __bf16* Bc, int k) -> v16bf {
        v8bf lo = *(const v8bf*)(Bc + k + 8 * hf);
        v8bf hi = *(const v8bf*)(Bc + k + 16 + 8 * hf);
        return __builtin_shufflevector(lo, hi, 0,1,2,3,4,5,6,7,
                                               8,9,10,11,12,13,14,15);
    };

    v8f acc0 = {}, acc1 = {};
    v16bf av = ldA(0), bv0 = ldB(B0, 0), bv1 = ldB(B1, 0);
    for (int k = 32; k < K; k += 32) {
        v16bf av2 = ldA(k), b02 = ldB(B0, k), b12 = ldB(B1, k);  // prefetch
        acc0 = __builtin_amdgcn_wmma_f32_16x16x32_bf16(false, av, false, bv0,
                                                       (short)0, acc0, false, false);
        acc1 = __builtin_amdgcn_wmma_f32_16x16x32_bf16(false, av, false, bv1,
                                                       (short)0, acc1, false, false);
        av = av2; bv0 = b02; bv1 = b12;
    }
    acc0 = __builtin_amdgcn_wmma_f32_16x16x32_bf16(false, av, false, bv0,
                                                   (short)0, acc0, false, false);
    acc1 = __builtin_amdgcn_wmma_f32_16x16x32_bf16(false, av, false, bv1,
                                                   (short)0, acc1, false, false);

    float* Crow = C + ((long long)b * M) * N;
#pragma unroll
    for (int p = 0; p < 8; ++p) {
        const int i = i0 + hf * 8 + p;
        Crow[(long long)i * N + (j0 + r)]      = acc0[p];
        Crow[(long long)i * N + (j0 + 16 + r)] = acc1[p];
    }
}

// ---------------------------------------------------------------------------
// SP branch: per-pixel channel softmax (64 ch) at 4 source pixels, per-channel
// max over the 2x2 window.  blockDim = 64 (one thread per class).
// ---------------------------------------------------------------------------
__global__ void softmax_pool_kern(const float* __restrict__ SP,
                                  float* __restrict__ SPf)
{
    const int c   = threadIdx.x;          // 0..63
    const int pix = blockIdx.x;           // 0..Mm-1
    const int b   = blockIdx.y;
    const int oy = pix / Nn, ox = pix % Nn;
    __shared__ float red[64];

    float best = -3.402823466e+38f;
#pragma unroll
    for (int dy = 0; dy < 2; ++dy)
#pragma unroll
        for (int dx = 0; dx < 2; ++dx) {
            const int y = oy * 2 + dy, x = ox * 2 + dx;
            const float v = SP[(((long long)b * Cs + c) * Hsp + y) * Hsp + x];
            red[c] = v; __syncthreads();
            for (int s = 32; s > 0; s >>= 1) {
                if (c < s) red[c] = fmaxf(red[c], red[c + s]);
                __syncthreads();
            }
            const float mx = red[0]; __syncthreads();
            const float e = __expf(v - mx);
            red[c] = e; __syncthreads();
            for (int s = 32; s > 0; s >>= 1) {
                if (c < s) red[c] += red[c + s];
                __syncthreads();
            }
            const float sum = red[0]; __syncthreads();
            best = fmaxf(best, e / sum);
        }
    SPf[((long long)b * Cs + c) * Mm + pix] = best;
}

// Row softmax of sp_adj -> bf16 (feeds the bf16 WMMA GEMM). 256 thr / row.
__global__ void row_softmax_bf16_kern(const float* __restrict__ A,
                                      __bf16* __restrict__ O, int M)
{
    const int row = blockIdx.x, tid = threadIdx.x;
    __shared__ float red[256];
    const float* r = A + (long long)row * M;

    float mx = -3.402823466e+38f;
    for (int j = tid; j < M; j += 256) mx = fmaxf(mx, r[j]);
    red[tid] = mx; __syncthreads();
    for (int s = 128; s > 0; s >>= 1) {
        if (tid < s) red[tid] = fmaxf(red[tid], red[tid + s]);
        __syncthreads();
    }
    mx = red[0]; __syncthreads();

    float sum = 0.f;
    for (int j = tid; j < M; j += 256) sum += __expf(r[j] - mx);
    red[tid] = sum; __syncthreads();
    for (int s = 128; s > 0; s >>= 1) {
        if (tid < s) red[tid] += red[tid + s];
        __syncthreads();
    }
    const float inv = 1.f / red[0];

    __bf16* o = O + (long long)row * M;
    for (int j = tid; j < M; j += 256)
        o[j] = (__bf16)(__expf(r[j] - mx) * inv);
}

// Fold BN params: inv = gamma*rsqrt(var+eps), shift = beta - mean*inv.
__global__ void bn_prep_kern(const float* g1, const float* b1, const float* m1,
                             const float* v1, const float* g2, const float* b2,
                             const float* m2, const float* v2,
                             float* i1, float* s1, float* i2, float* s2, int C)
{
    const int i = blockIdx.x * blockDim.x + threadIdx.x;
    if (i < C) {
        float iv = g1[i] * rsqrtf(v1[i] + EPSV);
        i1[i] = iv; s1[i] = b1[i] - m1[i] * iv;
        float jv = g2[i] * rsqrtf(v2[i] + EPSV);
        i2[i] = jv; s2[i] = b2[i] - m2[i] * jv;
    }
}

__global__ void cast_bf16_kern(const float* __restrict__ in,
                               __bf16* __restrict__ out, long long n)
{
    for (long long i = (long long)blockIdx.x * blockDim.x + threadIdx.x; i < n;
         i += (long long)gridDim.x * blockDim.x)
        out[i] = (__bf16)in[i];
}

// y3[b,c,m] = se[b,m,c] + sp[b,m,c] + t[b,c,m]   (token-major -> channel-major)
__global__ void fuse_y3_kern(const float* __restrict__ se,
                             const float* __restrict__ sp,
                             const float* __restrict__ t,
                             float* __restrict__ y3, long long n)
{
    for (long long idx = (long long)blockIdx.x * blockDim.x + threadIdx.x;
         idx < n; idx += (long long)gridDim.x * blockDim.x) {
        const int m = (int)(idx % Mm);
        const int c = (int)((idx / Mm) % Ci);
        const long long b = idx / ((long long)Mm * Ci);
        const long long tok = (b * Mm + m) * Ci + c;
        y3[idx] = se[tok] + sp[tok] + t[idx];
    }
}

// ---------------------------------------------------------------------------
static void gemm_f32(hipStream_t s,
                     const float* A, long long As0, long long As1, long long Ab,
                     const float* B, long long Bs0, long long Bs1, long long Bbt,
                     float* C, long long C0, long long C1, long long Cb,
                     const float* sm, const float* bm, const float* bn,
                     const float* res, long long R0, long long R1, long long Rb,
                     int M, int N, int K, int relu, int batch)
{
    dim3 grid((N + 63) / 64, M / 16, batch);
    dim3 block(128);
    const bool ac = (As1 == 1), bc = (Bs0 == 1);
    if (ac && bc)
        wmma_gemm_f32_kern<true, true><<<grid, block, 0, s>>>(
            A, As0, As1, Ab, B, Bs0, Bs1, Bbt, C, C0, C1, Cb,
            sm, bm, bn, res, R0, R1, Rb, M, N, K, relu);
    else if (ac)
        wmma_gemm_f32_kern<true, false><<<grid, block, 0, s>>>(
            A, As0, As1, Ab, B, Bs0, Bs1, Bbt, C, C0, C1, Cb,
            sm, bm, bn, res, R0, R1, Rb, M, N, K, relu);
    else if (bc)
        wmma_gemm_f32_kern<false, true><<<grid, block, 0, s>>>(
            A, As0, As1, Ab, B, Bs0, Bs1, Bbt, C, C0, C1, Cb,
            sm, bm, bn, res, R0, R1, Rb, M, N, K, relu);
    else
        wmma_gemm_f32_kern<false, false><<<grid, block, 0, s>>>(
            A, As0, As1, Ab, B, Bs0, Bs1, Bbt, C, C0, C1, Cb,
            sm, bm, bn, res, R0, R1, Rb, M, N, K, relu);
}

extern "C" void kernel_launch(void* const* d_in, const int* in_sizes, int n_in,
                              void* d_out, int out_size, void* d_ws, size_t ws_size,
                              hipStream_t stream)
{
    const float* x       = (const float*)d_in[0];
    const float* SP      = (const float*)d_in[1];
    const float* trans_w = (const float*)d_in[2];
    const float* bn1_g   = (const float*)d_in[3];
    const float* bn1_b   = (const float*)d_in[4];
    const float* bn1_m   = (const float*)d_in[5];
    const float* bn1_v   = (const float*)d_in[6];
    const float* linNC_w = (const float*)d_in[7];
    const float* linNC_b = (const float*)d_in[8];
    const float* linKC_w = (const float*)d_in[9];
    const float* linKC_b = (const float*)d_in[10];
    const float* gnn_w   = (const float*)d_in[11];
    const float* gnn_b   = (const float*)d_in[12];
    const float* sp_adj  = (const float*)d_in[13];
    const float* sp_w    = (const float*)d_in[14];
    const float* sp_b    = (const float*)d_in[15];
    const float* back_w  = (const float*)d_in[16];
    const float* bn2_g   = (const float*)d_in[17];
    const float* bn2_b   = (const float*)d_in[18];
    const float* bn2_m   = (const float*)d_in[19];
    const float* bn2_v   = (const float*)d_in[20];
    float* out = (float*)d_out;
    float* ws  = (float*)d_ws;

    size_t off = 0;
    auto alloc = [&](size_t nf) { size_t o = off; off += (nf + 63) & ~(size_t)63; return o; };

    float*  SPf  = ws + alloc((size_t)Bb * Cs * Mm);                 // (B,64,M)
    float*  t    = ws + alloc((size_t)Bb * Ci * Mm);                 // (B,128,M)
    __bf16* ybf  = (__bf16*)(ws + alloc(((size_t)Bb * Ci * Mm + 1) / 2));
    float*  yc   = ws + alloc((size_t)Bb * Ci * Cs);                 // (B,128,64)
    float*  sig  = ws + alloc((size_t)Bb * Cs * Cs);                 // (B,64,64)
    float*  P    = ws + alloc((size_t)Bb * Cs * Ci);                 // (B,64,128)
    float*  Q    = ws + alloc((size_t)Bb * Cs * Ci);                 // (B,64,128)
    float*  R    = ws + alloc((size_t)Bb * Mm * Ci);                 // (B,M,128)
    float*  se   = ws + alloc((size_t)Bb * Mm * Ci);                 // (B,M,128)
    __bf16* aspb = (__bf16*)(ws + alloc(((size_t)Mm * Mm + 1) / 2)); // (M,M)
    float*  spp  = ws + alloc((size_t)Bb * Mm * Ci);                 // (B,M,128)
    float*  bn1i = ws + alloc(Ci);
    float*  bn1s = ws + alloc(Ci);
    float*  bn2i = ws + alloc(Co);
    float*  bn2s = ws + alloc(Co);
    float*  spO  = R;    // reuse: R dead after `se`
    float*  y3   = spp;  // reuse: sp_pre dead after `sp`

    const long long tCM = (long long)Ci * Mm;   // t batch stride (channel-major)

    // 0) BN folds + SP branch + Asp softmax (independent front-end work)
    bn_prep_kern<<<1, 128, 0, stream>>>(bn1_g, bn1_b, bn1_m, bn1_v,
                                        bn2_g, bn2_b, bn2_m, bn2_v,
                                        bn1i, bn1s, bn2i, bn2s, Ci);
    softmax_pool_kern<<<dim3(Mm, Bb), 64, 0, stream>>>(SP, SPf);
    row_softmax_bf16_kern<<<Mm, 256, 0, stream>>>(sp_adj, aspb, Mm);

    // 1) t = relu(bn1(trans_w @ x))        (Ci x Cin) x (Cin x M) per batch
    gemm_f32(stream, trans_w, Cin, 1, 0,
             x, Mm, 1, (long long)Cin * Mm,
             t, Mm, 1, tCM,
             bn1i, bn1s, nullptr, nullptr, 0, 0, 0,
             Ci, Mm, Cin, 1, Bb);
    cast_bf16_kern<<<4096, 256, 0, stream>>>(t, ybf, (long long)Bb * Ci * Mm);

    // 2) yc = y @ linNC_w^T + b            (Ci x M) x (M x Cs)
    gemm_f32(stream, t, Mm, 1, tCM,
             linNC_w, 1, Mm, 0,
             yc, Cs, 1, (long long)Ci * Cs,
             nullptr, nullptr, linNC_b, nullptr, 0, 0, 0,
             Ci, Cs, Mm, 0, Bb);

    // 3) sigma = yc^T @ linKC_w^T + b      (Cs x Ci) x (Ci x Cs)
    gemm_f32(stream, yc, 1, Cs, (long long)Ci * Cs,
             linKC_w, 1, Ci, 0,
             sig, Cs, 1, (long long)Cs * Cs,
             nullptr, nullptr, linKC_b, nullptr, 0, 0, 0,
             Cs, Cs, Ci, 0, Bb);

    // Factored affinity: A@yT = SPf^T @ (sigma @ (SPf @ yT)) — never form A.
    // 4) P = SPf @ yT                      (Cs x M) x (M x Ci)
    gemm_f32(stream, SPf, Mm, 1, (long long)Cs * Mm,
             t, 1, Mm, tCM,
             P, Ci, 1, (long long)Cs * Ci,
             nullptr, nullptr, nullptr, nullptr, 0, 0, 0,
             Cs, Ci, Mm, 0, Bb);
    // 5) Q = sigma @ P                     (Cs x Cs) x (Cs x Ci)
    gemm_f32(stream, sig, Cs, 1, (long long)Cs * Cs,
             P, Ci, 1, (long long)Cs * Ci,
             Q, Ci, 1, (long long)Cs * Ci,
             nullptr, nullptr, nullptr, nullptr, 0, 0, 0,
             Cs, Ci, Cs, 0, Bb);
    // 6) R = SPf^T @ Q                     (M x Cs) x (Cs x Ci)
    gemm_f32(stream, SPf, 1, Mm, (long long)Cs * Mm,
             Q, Ci, 1, (long long)Cs * Ci,
             R, Ci, 1, (long long)Mm * Ci,
             nullptr, nullptr, nullptr, nullptr, 0, 0, 0,
             Mm, Ci, Cs, 0, Bb);
    // 7) se = relu(R @ gnn_w^T + gnn_b) + yT
    gemm_f32(stream, R, Ci, 1, (long long)Mm * Ci,
             gnn_w, 1, Ci, 0,
             se, Ci, 1, (long long)Mm * Ci,
             nullptr, nullptr, gnn_b, t, 1, Mm, tCM,
             Mm, Ci, Ci, 1, Bb);

    // 8) sp_pre = softmax(sp_adj) @ yT     (M x M bf16) x (M x Ci bf16) -> f32
    {
        dim3 grid((Ci + 127) / 128, Mm / 16, Bb);
        wmma_gemm_bf16_kern<<<grid, dim3(128), 0, stream>>>(
            aspb, ybf, spp, Mm, Ci, Mm);
    }
    // 9) sp = relu(sp_pre @ sp_w^T + sp_b) + yT   (reuses R's storage)
    gemm_f32(stream, spp, Ci, 1, (long long)Mm * Ci,
             sp_w, 1, Ci, 0,
             spO, Ci, 1, (long long)Mm * Ci,
             nullptr, nullptr, sp_b, t, 1, Mm, tCM,
             Mm, Ci, Ci, 1, Bb);

    // 10) y3[b,c,m] = se + sp (transpose) + t     (reuses sp_pre's storage)
    fuse_y3_kern<<<4096, 256, 0, stream>>>(se, spO, t, y3,
                                           (long long)Bb * Ci * Mm);

    // 11) out = relu(bn2(back_w @ y3))     (Co x Ci) x (Ci x M) per batch
    gemm_f32(stream, back_w, Ci, 1, 0,
             y3, Mm, 1, tCM,
             out, Mm, 1, (long long)Co * Mm,
             bn2i, bn2s, nullptr, nullptr, 0, 0, 0,
             Co, Mm, Ci, 1, Bb);
}
